// AttentionBase_86096914416430
// MI455X (gfx1250) — compile-verified, matrix operand order checked by reading, not run on hardware
//
#include <hip/hip_runtime.h>

#define B_ 2
#define S_ 2048
#define C_ 1024
#define H_ 16
#define D_ 64
#define KB 64   // keys per flash-attention iteration

typedef __attribute__((ext_vector_type(16))) _Float16 v16h;
typedef __attribute__((ext_vector_type(8)))  _Float16 v8h;
typedef __attribute__((ext_vector_type(8)))  float    v8f;
typedef __attribute__((ext_vector_type(4)))  int      v4i;

// ---------------- gfx1250 async global->LDS copy (16B per lane) ----------------
#if defined(__has_builtin)
#if __has_builtin(__builtin_amdgcn_global_load_async_to_lds_b128)
#define HAVE_ASYNC_LDS 1
#endif
#endif

#ifdef HAVE_ASYNC_LDS
typedef v4i __attribute__((address_space(1)))* gv4ip;  // global int4*
typedef v4i __attribute__((address_space(3)))* lv4ip;  // LDS int4*
#endif

__device__ __forceinline__ void async_copy16(const _Float16* gsrc, _Float16* ldst) {
#ifdef HAVE_ASYNC_LDS
  __builtin_amdgcn_global_load_async_to_lds_b128(
      (gv4ip)(void*)gsrc, (lv4ip)(void*)ldst, 0, 0);
#else
  *(uint4*)ldst = *(const uint4*)gsrc;
#endif
}

__device__ __forceinline__ void async_wait0() {
#ifdef HAVE_ASYNC_LDS
#if __has_builtin(__builtin_amdgcn_s_wait_asynccnt)
  __builtin_amdgcn_s_wait_asynccnt(0);
#else
  asm volatile("s_wait_asynccnt 0x0" ::: "memory");
#endif
#endif
}

// ---------------- lane reductions over 16-lane row groups ----------------
// xor-1/2/4 stay inside 8-lane groups -> v_mov_b32_dpp8 (pure VALU).
// xor-8 stays inside the 16-lane group -> single ds_swizzle_b32 immediate.
constexpr unsigned dpp8_sel_xor(unsigned m) {
  unsigned s = 0;
  for (unsigned i = 0; i < 8; ++i) s |= ((i ^ m) & 7u) << (3u * i);
  return s;
}

template <unsigned M>
__device__ __forceinline__ float xor_dpp8(float x) {
  constexpr int sel = (int)dpp8_sel_xor(M);
  return __builtin_bit_cast(float,
      __builtin_amdgcn_mov_dpp8(__builtin_bit_cast(int, x), sel));
}

__device__ __forceinline__ float xor8_swz(float x) {
  // ds_swizzle group-of-32: offset = {0, xor=8, or=0, and=0x1F} = 0x201F
  return __builtin_bit_cast(float,
      __builtin_amdgcn_ds_swizzle(__builtin_bit_cast(int, x), 0x201F));
}

__device__ __forceinline__ float rowmax16(float x) {
  x = fmaxf(x, xor_dpp8<1>(x));
  x = fmaxf(x, xor_dpp8<2>(x));
  x = fmaxf(x, xor_dpp8<4>(x));
  x = fmaxf(x, xor8_swz(x));
  return x;
}

__device__ __forceinline__ float rowsum16(float x) {
  x += xor_dpp8<1>(x);
  x += xor_dpp8<2>(x);
  x += xor_dpp8<4>(x);
  x += xor8_swz(x);
  return x;
}

// ---------------- fragment loaders ----------------
__device__ __forceinline__ void cvt8(v16h& f, int base, float4 a, float4 b, float sc) {
  f[base + 0] = (_Float16)(a.x * sc); f[base + 1] = (_Float16)(a.y * sc);
  f[base + 2] = (_Float16)(a.z * sc); f[base + 3] = (_Float16)(a.w * sc);
  f[base + 4] = (_Float16)(b.x * sc); f[base + 5] = (_Float16)(b.y * sc);
  f[base + 6] = (_Float16)(b.z * sc); f[base + 7] = (_Float16)(b.w * sc);
}

// 16x32 f16 A/B fragment row from row-major fp32 (CDNA5 ISA layout:
// lane-half h holds K = h*8..+7 in elems 0-7 and K = 16+h*8..+7 in elems 8-15)
__device__ __forceinline__ v16h gfrag_f32(const float* __restrict__ p, int lhal, float sc) {
  const float4* p4 = (const float4*)p;
  float4 a0 = p4[lhal * 2 + 0];
  float4 a1 = p4[lhal * 2 + 1];
  float4 b0 = p4[4 + lhal * 2 + 0];
  float4 b1 = p4[4 + lhal * 2 + 1];
  v16h f;
  cvt8(f, 0, a0, a1, sc);
  cvt8(f, 8, b0, b1, sc);
  return f;
}

// Same fragment from f16 memory (LDS or global); two 16B loads, no converts.
__device__ __forceinline__ v16h frag_f16(const _Float16* rowbase, int lhal) {
  const v8h* p = (const v8h*)rowbase;
  v8h lo = p[lhal];
  v8h hi = p[2 + lhal];
  v16h f;
#pragma unroll
  for (int e = 0; e < 8; ++e) { f[e] = lo[e]; f[8 + e] = hi[e]; }
  return f;
}

// -------------------- fused flash attention --------------------
// grid (S/64, H, B), block 128 = 4 wave32; each wave owns 16 query rows.
// Iterates over 64-key tiles staged in LDS (f16), online softmax in fp32.
__global__ void __launch_bounds__(128)
attn_fa_kernel(const float* __restrict__ q, const float* __restrict__ k,
               const float* __restrict__ v, const unsigned char* __restrict__ mask,
               _Float16* __restrict__ xout) {
  __shared__ __align__(16) _Float16 sK[KB][D_ + 8];     // key tile, row-major
  __shared__ __align__(16) _Float16 sVt[D_][KB + 8];    // value tile, transposed
  __shared__ __align__(16) _Float16 sP[4][16][KB + 8];  // per-wave P staging
  __shared__ float sBias[KB];

  const int qblk = blockIdx.x, h = blockIdx.y, b = blockIdx.z;
  const int tid  = threadIdx.x;
  const int wave = tid >> 5, lane = tid & 31;
  const int lhal = lane >> 4, lrow = lane & 15;

  const int q0 = qblk * 64 + wave * 16;
  const float* qp = q + ((long)b * S_ + q0 + lrow) * C_ + (long)h * D_;

  // Q fragments (16x64 as two K=32 chunks), pre-scaled by D^-1/2
  const v16h aq0 = gfrag_f32(qp, lhal, 0.125f);
  const v16h aq1 = gfrag_f32(qp + 32, lhal, 0.125f);

  float rm[8], rl[8];
  v8f acc[4];
#pragma unroll
  for (int i = 0; i < 8; ++i) { rm[i] = -3.0e38f; rl[i] = 0.0f; }
#pragma unroll
  for (int dg = 0; dg < 4; ++dg) {
    v8f z = {0.f, 0.f, 0.f, 0.f, 0.f, 0.f, 0.f, 0.f};
    acc[dg] = z;
  }

  const int sr  = tid >> 1;          // staging row  0..63
  const int scg = (tid & 1) * 32;    // staging cols {0,32}

  for (int kb = 0; kb < S_; kb += KB) {
    __syncthreads();  // previous tile fully consumed
    {
      const float4* kp4 = (const float4*)(k + ((long)b * S_ + kb + sr) * C_ + (long)h * D_ + scg);
      const float4* vp4 = (const float4*)(v + ((long)b * S_ + kb + sr) * C_ + (long)h * D_ + scg);
#pragma unroll
      for (int w = 0; w < 8; ++w) {
        float4 kf = kp4[w];
        sK[sr][scg + w * 4 + 0] = (_Float16)kf.x;
        sK[sr][scg + w * 4 + 1] = (_Float16)kf.y;
        sK[sr][scg + w * 4 + 2] = (_Float16)kf.z;
        sK[sr][scg + w * 4 + 3] = (_Float16)kf.w;
        float4 vf = vp4[w];
        sVt[scg + w * 4 + 0][sr] = (_Float16)vf.x;
        sVt[scg + w * 4 + 1][sr] = (_Float16)vf.y;
        sVt[scg + w * 4 + 2][sr] = (_Float16)vf.z;
        sVt[scg + w * 4 + 3][sr] = (_Float16)vf.w;
      }
      if (tid < KB) sBias[tid] = mask[(long)b * S_ + kb + tid] ? 0.0f : -1.0e12f;
      if (kb + KB < S_) {  // gfx1250 global_prefetch_b8 for next tile
        __builtin_prefetch(k + ((long)b * S_ + kb + KB + sr) * C_ + (long)h * D_ + scg, 0, 3);
        __builtin_prefetch(v + ((long)b * S_ + kb + KB + sr) * C_ + (long)h * D_ + scg, 0, 3);
      }
    }
    __syncthreads();

    // scores: S = (Q*scale) @ K^T for four 16-key groups (K-dim 64, chained)
    v8f s[4];
#pragma unroll
    for (int t = 0; t < 4; ++t) {
      const int key = t * 16 + lrow;
      v16h b0 = frag_f16(&sK[key][0], lhal);
      v16h b1 = frag_f16(&sK[key][32], lhal);
      v8f z = {0.f, 0.f, 0.f, 0.f, 0.f, 0.f, 0.f, 0.f};
      z    = __builtin_amdgcn_wmma_f32_16x16x32_f16(false, aq0, false, b0, (short)0, z, false, false);
      s[t] = __builtin_amdgcn_wmma_f32_16x16x32_f16(false, aq1, false, b1, (short)0, z, false, false);
    }

    // online softmax in fp32 (C-layout: row = i + 8*lhal, col = lrow)
    float bias[4];
#pragma unroll
    for (int t = 0; t < 4; ++t) bias[t] = sBias[t * 16 + lrow];

#pragma unroll
    for (int i = 0; i < 8; ++i) {
      float x0 = s[0][i] + bias[0];
      float x1 = s[1][i] + bias[1];
      float x2 = s[2][i] + bias[2];
      float x3 = s[3][i] + bias[3];
      float mx = rowmax16(fmaxf(fmaxf(x0, x1), fmaxf(x2, x3)));
      const float mnew = fmaxf(rm[i], mx);
      const float resc = __expf(rm[i] - mnew);
      rm[i] = mnew;
      const float p0 = __expf(x0 - mnew);
      const float p1 = __expf(x1 - mnew);
      const float p2 = __expf(x2 - mnew);
      const float p3 = __expf(x3 - mnew);
      const float rs = rowsum16((p0 + p1) + (p2 + p3));
      rl[i] = rl[i] * resc + rs;
#pragma unroll
      for (int dg = 0; dg < 4; ++dg) acc[dg][i] *= resc;
      const int m = i + 8 * lhal;
      sP[wave][m][lrow]      = (_Float16)p0;   // C-layout -> row-major P in LDS
      sP[wave][m][16 + lrow] = (_Float16)p1;
      sP[wave][m][32 + lrow] = (_Float16)p2;
      sP[wave][m][48 + lrow] = (_Float16)p3;
    }

    // O += P(16x64) @ V(64x64): P back in A-layout, V^T rows as B-fragments
    v16h pf0 = frag_f16(&sP[wave][lrow][0], lhal);
    v16h pf1 = frag_f16(&sP[wave][lrow][32], lhal);
#pragma unroll
    for (int dg = 0; dg < 4; ++dg) {
      v16h bv0 = frag_f16(&sVt[dg * 16 + lrow][0], lhal);
      v16h bv1 = frag_f16(&sVt[dg * 16 + lrow][32], lhal);
      acc[dg] = __builtin_amdgcn_wmma_f32_16x16x32_f16(false, pf0, false, bv0, (short)0,
                                                       acc[dg], false, false);
      acc[dg] = __builtin_amdgcn_wmma_f32_16x16x32_f16(false, pf1, false, bv1, (short)0,
                                                       acc[dg], false, false);
    }
  }

  // normalize, write attention output as f16 (projection consumes f16 anyway)
#pragma unroll
  for (int i = 0; i < 8; ++i) {
    const float inv = 1.0f / rl[i];
    _Float16* xr = xout + ((long)b * S_ + q0 + i + 8 * lhal) * C_ + (long)h * D_;
#pragma unroll
    for (int dg = 0; dg < 4; ++dg)
      xr[dg * 16 + lrow] = (_Float16)(acc[dg][i] * inv);
  }
}

// -------------------- output projection: Y = X @ W^T --------------------
// X is f16 [B*S, C] in workspace; W fp32 [C, C]; Y fp32. grid (M/64, N/64).
// Per K-step: X tile (64x32 f16) moves global->LDS via async-to-LDS (no VGPR
// round trip); W tile (64x32) staged once per block in f16 so all 4 waves
// share it (4x less W traffic + 4x fewer converts than per-wave loads).
__global__ void __launch_bounds__(128)
proj_kernel(const _Float16* __restrict__ X, const float* __restrict__ W,
            float* __restrict__ Y) {
  __shared__ __align__(16) _Float16 sX[64][32 + 8];
  __shared__ __align__(16) _Float16 sW[64][32 + 8];

  const int tid = threadIdx.x, wave = tid >> 5, lane = tid & 31;
  const int lhal = lane >> 4, lrow = lane & 15;
  const int m0blk = blockIdx.x * 64;
  const int n0 = blockIdx.y * 64;

  v8f acc[4];
#pragma unroll
  for (int nt = 0; nt < 4; ++nt) {
    v8f z = {0.f, 0.f, 0.f, 0.f, 0.f, 0.f, 0.f, 0.f};
    acc[nt] = z;
  }

  const int xc = (tid & 3) * 8;   // 16B chunk within a 32-half row
  const int xr = tid >> 2;        // rows xr and xr+32
  const int wn = tid >> 1;        // W row (output column) 0..63
  const int wk = (tid & 1) * 16;  // k half of the 32-wide chunk

  for (int kc = 0; kc < C_; kc += 32) {
    __syncthreads();  // previous tiles fully consumed

    // X tile: async global->LDS, 16B per lane, 2 rows per thread
    async_copy16(X + (long)(m0blk + xr) * C_ + kc + xc, &sX[xr][xc]);
    async_copy16(X + (long)(m0blk + xr + 32) * C_ + kc + xc, &sX[xr + 32][xc]);

    // W tile: fp32 -> f16 cooperative staging (shared by all 4 waves)
    {
      const float4* wp4 = (const float4*)(W + (long)(n0 + wn) * C_ + kc + wk);
#pragma unroll
      for (int w = 0; w < 4; ++w) {
        float4 wf = wp4[w];
        sW[wn][wk + w * 4 + 0] = (_Float16)wf.x;
        sW[wn][wk + w * 4 + 1] = (_Float16)wf.y;
        sW[wn][wk + w * 4 + 2] = (_Float16)wf.z;
        sW[wn][wk + w * 4 + 3] = (_Float16)wf.w;
      }
      if (kc + 32 < C_)
        __builtin_prefetch(W + (long)(n0 + wn) * C_ + kc + 32 + wk, 0, 3);
    }

    async_wait0();     // ASYNCcnt == 0: X tile resident in LDS
    __syncthreads();

    v16h af = frag_f16(&sX[wave * 16 + lrow][0], lhal);
#pragma unroll
    for (int nt = 0; nt < 4; ++nt) {
      v16h bf = frag_f16(&sW[nt * 16 + lrow][0], lhal);
      acc[nt] = __builtin_amdgcn_wmma_f32_16x16x32_f16(false, af, false, bf, (short)0,
                                                       acc[nt], false, false);
    }
  }

#pragma unroll
  for (int nt = 0; nt < 4; ++nt)
#pragma unroll
    for (int i = 0; i < 8; ++i)
      Y[(long)(m0blk + wave * 16 + i + 8 * lhal) * C_ + n0 + nt * 16 + lrow] = acc[nt][i];
}

extern "C" void kernel_launch(void* const* d_in, const int* in_sizes, int n_in,
                              void* d_out, int out_size, void* d_ws, size_t ws_size,
                              hipStream_t stream) {
  const float* q = (const float*)d_in[0];
  const float* k = (const float*)d_in[1];
  const float* v = (const float*)d_in[2];
  const float* W = (const float*)d_in[3];
  const unsigned char* mask = (const unsigned char*)d_in[4];  // bool [B,1,1,S]

  _Float16* X = (_Float16*)d_ws;  // attention output [B, S, C] f16 (8.4 MB)

  dim3 gA(S_ / 64, H_, B_);
  attn_fa_kernel<<<gA, dim3(128), 0, stream>>>(q, k, v, mask, X);

  dim3 gP((B_ * S_) / 64, C_ / 64);
  proj_kernel<<<gP, dim3(128), 0, stream>>>(X, W, (float*)d_out);
}
